// MambaBlock_37812892074508
// MI455X (gfx1250) — matrix operand order repaired and checked by reading, not compiled, not run on hardware
//
#include <hip/hip_runtime.h>

#define D_MODEL 1024
#define D_INNER 2048
#define D_STATE 16
#define D_CONV  4
#define DT_RANK 64
#define BATCH   2
#define SEQ     1024
#define NTOK    (BATCH * SEQ)            // 2048 tokens
#define XDBL_N  (DT_RANK + 2 * D_STATE)  // 96

typedef __attribute__((ext_vector_type(16))) __bf16 v16bf;
typedef __attribute__((ext_vector_type(8)))  __bf16 v8bf;
typedef __attribute__((ext_vector_type(8)))  float  v8f;

__device__ __forceinline__ unsigned short f2bf(float f) {
  unsigned int u = __float_as_uint(f);
  u += 0x7fffu + ((u >> 16) & 1u);       // round-to-nearest-even
  return (unsigned short)(u >> 16);
}

__device__ __forceinline__ float silu_f(float x) {
  return x / (1.f + __expf(-x));
}

union Q8 { uint4 q; v8bf h; };

// A fragment (16x32 bf16): lane half h holds K = h*8..h*8+7 and 16+h*8..+7
// p points at row_base + half*8 + kk (elements)
__device__ __forceinline__ v16bf load_a_frag(const unsigned short* p) {
  Q8 lo, hi;
  lo.q = *(const uint4*)(p);
  hi.q = *(const uint4*)(p + 16);
  return __builtin_shufflevector(lo.h, hi.h,
      0, 1, 2, 3, 4, 5, 6, 7, 8, 9, 10, 11, 12, 13, 14, 15);
}

// B fragment (32x16 bf16): lane half h holds K = h*8..h*8+7 and 16+h*8..+7
// p points at wrow_base + half*8 + kk (elements)
__device__ __forceinline__ v16bf load_b_frag(const unsigned short* p) {
  Q8 lo, hi;
  lo.q = *(const uint4*)(p);
  hi.q = *(const uint4*)(p + 16);
  return __builtin_shufflevector(lo.h, hi.h,
      0, 1, 2, 3, 4, 5, 6, 7, 8, 9, 10, 11, 12, 13, 14, 15);
}

// ---------------------------------------------------------------------------
// GEMM: C[M,N] = A[M,K](bf16, row-major, lda) * W[N,K](bf16, row-major, ldb)^T
// One wave per block; each wave computes a (16*MT) x (16*NT) tile of C.
// K must be a multiple of 64. Ping-pong double buffering (no register copies).
// mode 0: store; mode 1: + resid[row*ldc+col]; mode 2: softplus(v + bias[col])
// ---------------------------------------------------------------------------
template <int MT, int NT>
__global__ void __launch_bounds__(32)
gemm_bf16_wmma(const unsigned short* __restrict__ A, int lda,
               const unsigned short* __restrict__ W, int ldb,
               float* __restrict__ C, int ldc, int K,
               const float* __restrict__ resid,
               const float* __restrict__ bias, int mode) {
  const int lane = threadIdx.x;
  const int l16  = lane & 15;
  const int half = lane >> 4;
  const int m0   = blockIdx.y * (16 * MT);
  const int n0   = blockIdx.x * (16 * NT);

  const unsigned short* ap[MT];
#pragma unroll
  for (int i = 0; i < MT; ++i)
    ap[i] = A + (size_t)(m0 + i * 16 + l16) * lda + half * 8;
  const unsigned short* wp[NT];
#pragma unroll
  for (int j = 0; j < NT; ++j)
    wp[j] = W + (size_t)(n0 + j * 16 + l16) * ldb + half * 8;

  v8f acc[MT][NT];
#pragma unroll
  for (int i = 0; i < MT; ++i)
#pragma unroll
    for (int j = 0; j < NT; ++j)
#pragma unroll
      for (int r = 0; r < 8; ++r) acc[i][j][r] = 0.f;

  // ping-pong fragment buffers; stage 0 preloaded at k = 0
  v16bf a0[MT], b0[NT], a1[MT], b1[NT];
#pragma unroll
  for (int i = 0; i < MT; ++i) a0[i] = load_a_frag(ap[i]);
#pragma unroll
  for (int j = 0; j < NT; ++j) b0[j] = load_b_frag(wp[j]);

  for (int kk = 0; kk < K; kk += 64) {
    // stage-1 fragments at kk+32 (always in range: K % 64 == 0)
#pragma unroll
    for (int i = 0; i < MT; ++i) a1[i] = load_a_frag(ap[i] + kk + 32);
#pragma unroll
    for (int j = 0; j < NT; ++j) b1[j] = load_b_frag(wp[j] + kk + 32);
    if (kk + 128 < K) __builtin_prefetch(ap[0] + kk + 128, 0, 1);

#pragma unroll
    for (int i = 0; i < MT; ++i)
#pragma unroll
      for (int j = 0; j < NT; ++j)
        acc[i][j] = __builtin_amdgcn_wmma_f32_16x16x32_bf16(
            false, a0[i], false, b0[j], (short)0, acc[i][j], false, false);

    if (kk + 64 < K) {
#pragma unroll
      for (int i = 0; i < MT; ++i) a0[i] = load_a_frag(ap[i] + kk + 64);
#pragma unroll
      for (int j = 0; j < NT; ++j) b0[j] = load_b_frag(wp[j] + kk + 64);
    }

#pragma unroll
    for (int i = 0; i < MT; ++i)
#pragma unroll
      for (int j = 0; j < NT; ++j)
        acc[i][j] = __builtin_amdgcn_wmma_f32_16x16x32_bf16(
            false, a1[i], false, b1[j], (short)0, acc[i][j], false, false);
  }

#pragma unroll
  for (int i = 0; i < MT; ++i) {
#pragma unroll
    for (int j = 0; j < NT; ++j) {
      const int col = n0 + j * 16 + l16;
#pragma unroll
      for (int r = 0; r < 8; ++r) {
        const int row = m0 + i * 16 + half * 8 + r;
        float v = acc[i][j][r];
        if (mode == 1) {
          v += resid[(size_t)row * ldc + col];
        } else if (mode == 2) {
          v += bias[col];
          v = (v > 20.f) ? v : log1pf(__expf(v));   // softplus
        }
        C[(size_t)row * ldc + col] = v;
      }
    }
  }
}

// ---------------------------------------------------------------------------
// RMSNorm over D_MODEL, output bf16. One block (256 thr) per token.
// ---------------------------------------------------------------------------
__global__ void rmsnorm_bf16_kernel(const float* __restrict__ x,
                                    const float* __restrict__ w,
                                    unsigned short* __restrict__ h) {
  const int t = blockIdx.x;
  const float* row = x + (size_t)t * D_MODEL;
  float s = 0.f;
  for (int i = threadIdx.x; i < D_MODEL; i += 256) {
    float v = row[i];
    s = fmaf(v, v, s);
  }
#pragma unroll
  for (int off = 16; off > 0; off >>= 1) s += __shfl_down(s, off, 32);
  __shared__ float partial[8];
  __shared__ float rs_sh;
  if ((threadIdx.x & 31) == 0) partial[threadIdx.x >> 5] = s;
  __syncthreads();
  if (threadIdx.x == 0) {
    float tsum = 0.f;
#pragma unroll
    for (int i = 0; i < 8; ++i) tsum += partial[i];
    rs_sh = rsqrtf(tsum / (float)D_MODEL + 1e-5f);
  }
  __syncthreads();
  const float rs = rs_sh;
  unsigned short* hr = h + (size_t)t * D_MODEL;
  for (int i = threadIdx.x; i < D_MODEL; i += 256)
    hr[i] = f2bf(row[i] * rs * w[i]);
}

__global__ void f32_to_bf16_kernel(const float* __restrict__ s,
                                   unsigned short* __restrict__ d, int n) {
  int i = blockIdx.x * 256 + threadIdx.x;
  if (i < n) d[i] = f2bf(s[i]);
}

// ---------------------------------------------------------------------------
// Causal depthwise conv (width 4) + bias + SiLU. One thread per (token, chan).
// xr = [NTOK, 2*D_INNER]; x-path is cols [0, D_INNER).
// ---------------------------------------------------------------------------
__global__ void conv_silu_kernel(const float* __restrict__ xr,
                                 const float* __restrict__ cw,
                                 const float* __restrict__ cb,
                                 float* __restrict__ u,
                                 unsigned short* __restrict__ u_bf) {
  int idx = blockIdx.x * 256 + threadIdx.x;        // over NTOK * D_INNER
  int c = idx & (D_INNER - 1);
  int t = idx >> 11;
  int b = t >> 10, l = t & (SEQ - 1);
  float acc = cb[c];
#pragma unroll
  for (int k = 0; k < D_CONV; ++k) {
    int li = l - (D_CONV - 1) + k;
    if (li >= 0)
      acc = fmaf(xr[(size_t)(b * SEQ + li) * (2 * D_INNER) + c], cw[c * D_CONV + k], acc);
  }
  float s = silu_f(acc);
  u[idx]    = s;
  u_bf[idx] = f2bf(s);
}

// ---------------------------------------------------------------------------
// Selective scan. One lane per (batch, channel) -> 4096 lanes, 16 states in
// registers each. B[t][n]/C[t][n] come from x_dbl via one lane-load + shuffles.
// ---------------------------------------------------------------------------
__global__ void selective_scan_kernel(const float* __restrict__ delta,
                                      const float* __restrict__ u,
                                      const float* __restrict__ xdbl,
                                      const float* __restrict__ A_log,
                                      const float* __restrict__ Dv,
                                      float* __restrict__ ypre) {
  int g = blockIdx.x * 256 + threadIdx.x;   // 0 .. BATCH*D_INNER-1
  int b = g >> 11;
  int d = g & (D_INNER - 1);
  int lane = threadIdx.x & 31;

  float Arow[D_STATE];
#pragma unroll
  for (int n = 0; n < D_STATE; ++n) Arow[n] = -__expf(A_log[d * D_STATE + n]);
  const float Dd = Dv[d];

  float st[D_STATE];
#pragma unroll
  for (int n = 0; n < D_STATE; ++n) st[n] = 0.f;

  for (int l = 0; l < SEQ; ++l) {
    const int t = b * SEQ + l;
    const float dt = delta[(size_t)t * D_INNER + d];
    const float ut = u[(size_t)t * D_INNER + d];
    // lanes 0..15 read B[t][lane], lanes 16..31 read C[t][lane-16]
    const float bc = xdbl[(size_t)t * XDBL_N + DT_RANK + lane];
    const float du = dt * ut;
    float y = 0.f;
#pragma unroll
    for (int n = 0; n < D_STATE; ++n) {
      float Bn = __shfl(bc, n, 32);
      float Cn = __shfl(bc, n + 16, 32);
      float sA = __expf(dt * Arow[n]);
      st[n] = fmaf(sA, st[n], du * Bn);
      y = fmaf(st[n], Cn, y);
    }
    ypre[(size_t)t * D_INNER + d] = fmaf(ut, Dd, y);
  }
}

// y_gated = ypre * silu(res), cast to bf16 for the output GEMM.
__global__ void gate_kernel(const float* __restrict__ ypre,
                            const float* __restrict__ xr,
                            unsigned short* __restrict__ yg) {
  int idx = blockIdx.x * 256 + threadIdx.x;        // over NTOK * D_INNER
  int c = idx & (D_INNER - 1);
  int t = idx >> 11;
  float r = xr[(size_t)t * (2 * D_INNER) + D_INNER + c];
  yg[idx] = f2bf(ypre[idx] * silu_f(r));
}

// ---------------------------------------------------------------------------
extern "C" void kernel_launch(void* const* d_in, const int* in_sizes, int n_in,
                              void* d_out, int out_size, void* d_ws, size_t ws_size,
                              hipStream_t stream) {
  (void)in_sizes; (void)n_in; (void)out_size; (void)ws_size;
  const float* x       = (const float*)d_in[0];
  const float* norm_w  = (const float*)d_in[1];
  const float* in_w    = (const float*)d_in[2];
  const float* conv_w  = (const float*)d_in[3];
  const float* conv_b  = (const float*)d_in[4];
  const float* xproj_w = (const float*)d_in[5];
  const float* dt_w    = (const float*)d_in[6];
  const float* dt_b    = (const float*)d_in[7];
  const float* A_log   = (const float*)d_in[8];
  const float* Dv      = (const float*)d_in[9];
  const float* out_w   = (const float*)d_in[10];
  float* out = (float*)d_out;

  char* p = (char*)d_ws;
  auto take = [&](size_t bytes) {
    void* r = (void*)p;
    p += (bytes + 255) & ~(size_t)255;
    return r;
  };
  unsigned short* h_bf    = (unsigned short*)take((size_t)NTOK * D_MODEL * 2);
  unsigned short* win_bf  = (unsigned short*)take((size_t)2 * D_INNER * D_MODEL * 2);
  unsigned short* wxp_bf  = (unsigned short*)take((size_t)XDBL_N * D_INNER * 2);
  unsigned short* wdt_bf  = (unsigned short*)take((size_t)D_INNER * DT_RANK * 2);
  unsigned short* wout_bf = (unsigned short*)take((size_t)D_MODEL * D_INNER * 2);
  float*          xr      = (float*)take((size_t)NTOK * 2 * D_INNER * 4);
  float*          u       = (float*)take((size_t)NTOK * D_INNER * 4);
  unsigned short* u_bf    = (unsigned short*)take((size_t)NTOK * D_INNER * 2);
  float*          xdbl    = (float*)take((size_t)NTOK * XDBL_N * 4);
  unsigned short* xdbl_bf = (unsigned short*)take((size_t)NTOK * XDBL_N * 2);
  float*          delta   = (float*)take((size_t)NTOK * D_INNER * 4);
  float*          ypre    = (float*)take((size_t)NTOK * D_INNER * 4);
  unsigned short* yg_bf   = (unsigned short*)take((size_t)NTOK * D_INNER * 2);

  // 1. RMSNorm -> bf16 activations
  rmsnorm_bf16_kernel<<<NTOK, 256, 0, stream>>>(x, norm_w, h_bf);

  // 2. Weight casts (f32 -> bf16)
  {
    int n;
    n = 2 * D_INNER * D_MODEL;
    f32_to_bf16_kernel<<<(n + 255) / 256, 256, 0, stream>>>(in_w, win_bf, n);
    n = XDBL_N * D_INNER;
    f32_to_bf16_kernel<<<(n + 255) / 256, 256, 0, stream>>>(xproj_w, wxp_bf, n);
    n = D_INNER * DT_RANK;
    f32_to_bf16_kernel<<<(n + 255) / 256, 256, 0, stream>>>(dt_w, wdt_bf, n);
    n = D_MODEL * D_INNER;
    f32_to_bf16_kernel<<<(n + 255) / 256, 256, 0, stream>>>(out_w, wout_bf, n);
  }

  // 3. in_proj: xr[2048,4096] = h * Win^T   (32x64 tiles per wave)
  gemm_bf16_wmma<2, 4><<<dim3((2 * D_INNER) / 64, NTOK / 32), 32, 0, stream>>>(
      h_bf, D_MODEL, win_bf, D_MODEL, xr, 2 * D_INNER, D_MODEL,
      nullptr, nullptr, 0);

  // 4. causal depthwise conv + SiLU -> u (f32 + bf16)
  conv_silu_kernel<<<(NTOK * D_INNER) / 256, 256, 0, stream>>>(xr, conv_w, conv_b, u, u_bf);

  // 5. x_proj: xdbl[2048,96] = u * Wxp^T  (N=96 -> NT=3, grid.x=2)
  gemm_bf16_wmma<2, 3><<<dim3(2, NTOK / 32), 32, 0, stream>>>(
      u_bf, D_INNER, wxp_bf, D_INNER, xdbl, XDBL_N, D_INNER,
      nullptr, nullptr, 0);

  // 6. cast x_dbl for dt GEMM
  {
    int n = NTOK * XDBL_N;
    f32_to_bf16_kernel<<<(n + 255) / 256, 256, 0, stream>>>(xdbl, xdbl_bf, n);
  }

  // 7. dt_proj + bias + softplus: delta[2048,2048] (A = xdbl[:, :64], lda=96, K=64)
  gemm_bf16_wmma<2, 4><<<dim3(D_INNER / 64, NTOK / 32), 32, 0, stream>>>(
      xdbl_bf, XDBL_N, wdt_bf, DT_RANK, delta, D_INNER, DT_RANK,
      nullptr, dt_b, 2);

  // 8. selective scan (register-resident states, shuffle-broadcast B/C)
  selective_scan_kernel<<<(BATCH * D_INNER) / 256, 256, 0, stream>>>(
      delta, u, xdbl, A_log, Dv, ypre);

  // 9. gate: yg = ypre * silu(res) -> bf16
  gate_kernel<<<(NTOK * D_INNER) / 256, 256, 0, stream>>>(ypre, xr, yg_bf);

  // 10. out_proj + residual: out[2048,1024] = yg * Wout^T + x
  gemm_bf16_wmma<2, 4><<<dim3(D_MODEL / 64, NTOK / 32), 32, 0, stream>>>(
      yg_bf, D_INNER, wout_bf, D_INNER, out, D_MODEL, D_INNER,
      x, nullptr, 1);
}
